// MeshMultiHeadHodgeLaplaceAttention_6408091206160
// MI455X (gfx1250) — matrix operand order repaired and checked by reading, not compiled, not run on hardware
//
#include <hip/hip_runtime.h>

// ---------------------------------------------------------------------------
// MeshMultiHeadHodgeLaplaceAttention for MI455X (gfx1250, wave32, WMMA).
// One generic batched bf16-WMMA GEMM (fp32 accumulate) with:
//   - BN = 256 = full N  -> each d_0/d_1 element streamed from HBM once/pass
//   - compile-time TRANSA (no per-iteration branches), peeled last k-step
//   - single prefetch register set: store->barrier->load(next)->compute
//   - packed fp32->bf16x2 conversion, dword LDS stores
//   - fused per-(row,head) Hodge-diagonal epilogues and branch-sum add
// Fragment layouts per CDNA5 ISA §7.12.2:
//   A 16x32 16-bit: lane(r,half): elems 0..7 = K[half*8..+7],
//                                 elems 8..15 = K[16+half*8..+7]   (interleaved)
//   B 32x16 16-bit: lane(c,half): elems 0..7 = K[half*8..+7], 8..15 = K[16+half*8..+7] (interleaved)
//   C/D 16x16 f32 : lane(r,half): vgpr v = C[half*8 + v][r]
// ---------------------------------------------------------------------------

typedef __bf16 bf16_t;
typedef __attribute__((ext_vector_type(2)))  __bf16 v2bf;
typedef __attribute__((ext_vector_type(8)))  __bf16 v8bf;
typedef __attribute__((ext_vector_type(16))) __bf16 v16bf;
typedef __attribute__((ext_vector_type(8)))  float  v8f;

#define BM 64
#define BN 256
#define BK 32
#define LDSTR 40           // bf16 units per LDS row (80B; 16B-aligned frag loads)
#define THREADS 256        // 8 wave32s: 2 (M) x 4 (N), wave tile 32x64

// pack two floats into a dword of 2 bf16
static __device__ __forceinline__ unsigned pk2bf(float a, float b) {
#if __has_builtin(__builtin_amdgcn_cvt_pk_bf16_f32)
    v2bf v = __builtin_amdgcn_cvt_pk_bf16_f32(a, b);
    unsigned u; __builtin_memcpy(&u, &v, 4);
    return u;
#else
    // native truncating casts: lets ISel pick v_cvt_(pk_)bf16_f32 if present
    v2bf v; v[0] = (__bf16)a; v[1] = (__bf16)b;
    unsigned u; __builtin_memcpy(&u, &v, 4);
    return u;
#endif
}

// C[b] = op(A[b]) x B[b], fp32 in/out, bf16 WMMA inside.
//   TRANSA=false: A is M x K row-major (lda = row stride)
//   TRANSA=true : op(A)[i,k] = A[k*lda + i]  (original A is K x lda row-major)
// Epilogue: scaleMode 0: none; 1: *= s[(b*8+col/32)*srows+row];
//           2: *= 1/(s[..]+1e-6). Optional addP: += addP[b*strideAdd+row*ldc+col].
template <bool TRANSA>
__global__ __launch_bounds__(THREADS)
void gemm_bf16_wmma(const float* __restrict__ A, const float* __restrict__ B,
                    float* __restrict__ C,
                    int M, int N, int K,
                    int lda, int ldb, int ldc,
                    long long strideA, long long strideB, long long strideC,
                    const float* __restrict__ scaleP, int srows, int scaleMode,
                    const float* __restrict__ addP, long long strideAdd)
{
    __shared__ bf16_t As[BM * LDSTR];   //  5.0 KB
    __shared__ bf16_t Bs[BN * LDSTR];   // 20.0 KB

    const int batch = blockIdx.z;
    const float* Ag = A + (long long)batch * strideA;
    const float* Bg = B + (long long)batch * strideB;
    float*       Cg = C + (long long)batch * strideC;

    const int tid  = threadIdx.x;
    const int lane = tid & 31;
    const int wave = tid >> 5;
    const int wm   = wave >> 2;     // 0..1  (wave row: 32 rows)
    const int wn   = wave & 3;      // 0..3  (wave col: 64 cols)
    const int r    = lane & 15;
    const int half = lane >> 4;

    const int m0 = blockIdx.y * BM;
    const int n0 = blockIdx.x * BN;

    // staging coordinates (fixed across k-steps)
    const int ta_kk2 = tid >> 4;          // TRANSA: k-pair 0..15
    const int ta_rr4 = (tid & 15) << 2;   // TRANSA: row group

    v8f acc[2][4];
    #pragma unroll
    for (int mi = 0; mi < 2; ++mi)
        #pragma unroll
        for (int ni = 0; ni < 4; ++ni)
            acc[mi][ni] = v8f{0.f,0.f,0.f,0.f,0.f,0.f,0.f,0.f};

    float4 pa[2];     // A prefetch (8 floats)
    float4 pb[8];     // B prefetch (32 floats)

    auto loadTiles = [&](int k0) {
        if constexpr (!TRANSA) {
            #pragma unroll
            for (int it = 0; it < 2; ++it) {
                int id  = tid + it * THREADS;
                int row = id >> 3;
                int kk  = (id & 7) << 2;
                pa[it] = *(const float4*)(Ag + (long long)(m0 + row) * lda + (k0 + kk));
            }
        } else {
            pa[0] = *(const float4*)(Ag + (long long)(k0 + 2*ta_kk2 + 0) * lda + (m0 + ta_rr4));
            pa[1] = *(const float4*)(Ag + (long long)(k0 + 2*ta_kk2 + 1) * lda + (m0 + ta_rr4));
        }
        #pragma unroll
        for (int it = 0; it < 4; ++it) {
            int id  = tid + it * THREADS;
            int kk2 = id >> 6;
            int nn4 = (id & 63) << 2;
            pb[2*it + 0] = *(const float4*)(Bg + (long long)(k0 + 2*kk2 + 0) * ldb + (n0 + nn4));
            pb[2*it + 1] = *(const float4*)(Bg + (long long)(k0 + 2*kk2 + 1) * ldb + (n0 + nn4));
        }
    };

    auto storeTiles = [&]() {
        if constexpr (!TRANSA) {
            #pragma unroll
            for (int it = 0; it < 2; ++it) {
                int id  = tid + it * THREADS;
                int row = id >> 3;
                int kk  = (id & 7) << 2;
                uint2 u;
                u.x = pk2bf(pa[it].x, pa[it].y);
                u.y = pk2bf(pa[it].z, pa[it].w);
                *(uint2*)&As[row * LDSTR + kk] = u;
            }
        } else {
            const float* f0 = (const float*)&pa[0];
            const float* f1 = (const float*)&pa[1];
            #pragma unroll
            for (int i = 0; i < 4; ++i)
                *(unsigned*)&As[(ta_rr4 + i) * LDSTR + 2*ta_kk2] = pk2bf(f0[i], f1[i]);
        }
        #pragma unroll
        for (int it = 0; it < 4; ++it) {
            int id  = tid + it * THREADS;
            int kk2 = id >> 6;
            int nn4 = (id & 63) << 2;
            const float* f0 = (const float*)&pb[2*it + 0];
            const float* f1 = (const float*)&pb[2*it + 1];
            #pragma unroll
            for (int i = 0; i < 4; ++i)
                *(unsigned*)&Bs[(nn4 + i) * LDSTR + 2*kk2] = pk2bf(f0[i], f1[i]);
        }
    };

    auto compute = [&]() {
        // B fragment: interleaved K quarters per half-wave (matches A)
        v16bf bfr[4];
        #pragma unroll
        for (int ni = 0; ni < 4; ++ni) {
            int col = wn * 64 + ni * 16 + r;
            const v8bf blo = *(const v8bf*)&Bs[col * LDSTR + half * 8];
            const v8bf bhi = *(const v8bf*)&Bs[col * LDSTR + 16 + half * 8];
            bfr[ni] = __builtin_shufflevector(blo, bhi, 0,1,2,3,4,5,6,7,8,9,10,11,12,13,14,15);
        }
        // A fragment: interleaved K quarters per half-wave (ISA A-layout)
        #pragma unroll
        for (int mi = 0; mi < 2; ++mi) {
            int row = wm * 32 + mi * 16 + r;
            const v8bf lo = *(const v8bf*)&As[row * LDSTR + half * 8];
            const v8bf hi = *(const v8bf*)&As[row * LDSTR + 16 + half * 8];
            v16bf afr = __builtin_shufflevector(lo, hi, 0,1,2,3,4,5,6,7,8,9,10,11,12,13,14,15);
            #pragma unroll
            for (int ni = 0; ni < 4; ++ni)
                acc[mi][ni] = __builtin_amdgcn_wmma_f32_16x16x32_bf16(
                    false, afr, false, bfr[ni], (short)0, acc[mi][ni], false, false);
        }
    };

    loadTiles(0);
    int k0 = 0;
    for (; k0 + BK < K; k0 += BK) {
        storeTiles();               // consumes pa/pb
        __syncthreads();
        loadTiles(k0 + BK);         // refill pa/pb; overlaps HBM latency with WMMA
        compute();
        __syncthreads();
    }
    storeTiles();                   // peeled last step: no predicate in hot loop
    __syncthreads();
    compute();

    // ------- epilogue: Hodge-diagonal scale (+ optional add), fp32 store -------
    #pragma unroll
    for (int mi = 0; mi < 2; ++mi) {
        #pragma unroll
        for (int ni = 0; ni < 4; ++ni) {
            const int gcol = n0 + wn * 64 + ni * 16 + r;
            const int h    = gcol >> 5;                 // head index (d_head = 32)
            const int rowb = m0 + wm * 32 + mi * 16 + half * 8;

            float sv[8];
            if (scaleMode != 0) {
                const float* sp = scaleP + (long long)(batch * 8 + h) * srows + rowb;
                float4 s0 = *(const float4*)(sp);       // rows rowb..rowb+3
                float4 s1 = *(const float4*)(sp + 4);   // rows rowb+4..rowb+7
                sv[0]=s0.x; sv[1]=s0.y; sv[2]=s0.z; sv[3]=s0.w;
                sv[4]=s1.x; sv[5]=s1.y; sv[6]=s1.z; sv[7]=s1.w;
            }
            #pragma unroll
            for (int vi = 0; vi < 8; ++vi) {
                float val = acc[mi][ni][vi];
                if (scaleMode == 1)      val *= sv[vi];
                else if (scaleMode == 2) val *= 1.0f / (sv[vi] + 1e-6f);
                long long off = (long long)(rowb + vi) * ldc + gcol;
                if (addP) val += addP[(long long)batch * strideAdd + off];
                Cg[off] = val;
            }
        }
    }
}

// hval[b,h,i] = dot(Q[row, h*32..], K[row, h*32..]) / sqrt(32)
__global__ void rowdot_hval(const float* __restrict__ Q, const float* __restrict__ Kp,
                            float* __restrict__ hval, int rowsPerBatch, int total)
{
    int t = blockIdx.x * blockDim.x + threadIdx.x;
    if (t >= total) return;
    int row = t >> 3;
    int h   = t & 7;
    const float* q = Q  + (long long)row * 256 + h * 32;
    const float* k = Kp + (long long)row * 256 + h * 32;
    float s = 0.f;
    #pragma unroll
    for (int i = 0; i < 32; ++i) s += q[i] * k[i];
    int b = row / rowsPerBatch;
    int i = row - b * rowsPerBatch;
    hval[(long long)(b * 8 + h) * rowsPerBatch + i] = s * 0.17677669529663687f; // 1/sqrt(32)
}

// Y[b,i,col] = X[b,i,col] * S[b, col/32, i]
__global__ void scale_rows(const float* __restrict__ X, const float* __restrict__ S,
                           float* __restrict__ Y, int rowsPerBatch, long long total)
{
    long long idx = (long long)blockIdx.x * blockDim.x + threadIdx.x;
    if (idx >= total) return;
    int col = (int)(idx & 255);
    long long row = idx >> 8;
    int b = (int)(row / rowsPerBatch);
    int i = (int)(row - (long long)b * rowsPerBatch);
    Y[idx] = X[idx] * S[(long long)(b * 8 + (col >> 5)) * rowsPerBatch + i];
}

extern "C" void kernel_launch(void* const* d_in, const int* in_sizes, int n_in,
                              void* d_out, int out_size, void* d_ws, size_t ws_size,
                              hipStream_t stream) {
    const float* x_v  = (const float*)d_in[0];
    const float* x_e  = (const float*)d_in[1];
    const float* x_f  = (const float*)d_in[2];
    const float* d_0  = (const float*)d_in[3];
    const float* d_1  = (const float*)d_in[4];
    const float* W_vq = (const float*)d_in[5];
    const float* W_vk = (const float*)d_in[6];
    const float* W_eq = (const float*)d_in[7];
    const float* W_ek = (const float*)d_in[8];
    const float* W_eiq= (const float*)d_in[9];
    const float* W_eik= (const float*)d_in[10];
    const float* W_ev = (const float*)d_in[11];
    const float* W_fq = (const float*)d_in[12];
    const float* W_fk = (const float*)d_in[13];
    const float* W_o  = (const float*)d_in[14];
    float* out = (float*)d_out;

    constexpr int Bb = 2, Nv = 2048, Me = 6144, Kf = 4096, D = 256;

    // workspace carve-up (fp32)
    float* w    = (float*)d_ws;
    float* qtmp = w; w += (long long)Bb * Me * D;   // max proj rows = 12288
    float* ktmp = w; w += (long long)Bb * Me * D;
    float* hv   = w; w += (long long)Bb * 8 * Nv;
    float* he   = w; w += (long long)Bb * 8 * Me;
    float* hei  = w; w += (long long)Bb * 8 * Me;
    float* hf   = w; w += (long long)Bb * 8 * Kf;
    float* eV   = w; w += (long long)Bb * Me * D;
    float* t1   = w; w += (long long)Bb * Kf * D;
    float* t2   = w; w += (long long)Bb * Me * D;
    float* y0   = w; w += (long long)Bb * Me * D;   // reused for (y2 + t2)
    float* y1   = w; w += (long long)Bb * Nv * D;

    auto gemm = [&](const float* A, const float* Bm, float* C, int M, int N, int K,
                    int lda, int ldb, int ldc,
                    long long sA, long long sB, long long sC, int nb, int transA,
                    const float* sp, int srows, int smode,
                    const float* addP, long long sAdd) {
        dim3 grid(N / BN, M / BM, nb);
        if (transA)
            gemm_bf16_wmma<true><<<grid, dim3(THREADS), 0, stream>>>(
                A, Bm, C, M, N, K, lda, ldb, ldc, sA, sB, sC,
                sp, srows, smode, addP, sAdd);
        else
            gemm_bf16_wmma<false><<<grid, dim3(THREADS), 0, stream>>>(
                A, Bm, C, M, N, K, lda, ldb, ldc, sA, sB, sC,
                sp, srows, smode, addP, sAdd);
    };

    // ---- Hodge diagonals via Q/K projections + rowwise dot ----
    gemm(x_v, W_vq, qtmp, Bb*Nv, D, D, D, D, D, 0,0,0, 1, 0, nullptr,0,0, nullptr,0);
    gemm(x_v, W_vk, ktmp, Bb*Nv, D, D, D, D, D, 0,0,0, 1, 0, nullptr,0,0, nullptr,0);
    rowdot_hval<<<(Bb*Nv*8 + 255)/256, 256, 0, stream>>>(qtmp, ktmp, hv, Nv, Bb*Nv*8);

    gemm(x_e, W_eq, qtmp, Bb*Me, D, D, D, D, D, 0,0,0, 1, 0, nullptr,0,0, nullptr,0);
    gemm(x_e, W_ek, ktmp, Bb*Me, D, D, D, D, D, 0,0,0, 1, 0, nullptr,0,0, nullptr,0);
    rowdot_hval<<<(Bb*Me*8 + 255)/256, 256, 0, stream>>>(qtmp, ktmp, he, Me, Bb*Me*8);

    gemm(x_e, W_eiq, qtmp, Bb*Me, D, D, D, D, D, 0,0,0, 1, 0, nullptr,0,0, nullptr,0);
    gemm(x_e, W_eik, ktmp, Bb*Me, D, D, D, D, D, 0,0,0, 1, 0, nullptr,0,0, nullptr,0);
    rowdot_hval<<<(Bb*Me*8 + 255)/256, 256, 0, stream>>>(qtmp, ktmp, hei, Me, Bb*Me*8);

    gemm(x_f, W_fq, qtmp, Bb*Kf, D, D, D, D, D, 0,0,0, 1, 0, nullptr,0,0, nullptr,0);
    gemm(x_f, W_fk, ktmp, Bb*Kf, D, D, D, D, D, 0,0,0, 1, 0, nullptr,0,0, nullptr,0);
    rowdot_hval<<<(Bb*Kf*8 + 255)/256, 256, 0, stream>>>(qtmp, ktmp, hf, Kf, Bb*Kf*8);

    // ---- e_V projection ----
    gemm(x_e, W_ev, eV, Bb*Me, D, D, D, D, D, 0,0,0, 1, 0, nullptr,0,0, nullptr,0);

    // ---- up-Laplacian branch: *1^{-1} d_1^T *2 d_1 ----
    gemm(d_1, eV, t1, Kf, D, Me, Me, D, D,
         (long long)Kf*Me, (long long)Me*D, (long long)Kf*D, Bb, 0,
         hf, Kf, 1, nullptr, 0);
    gemm(d_1, t1, t2, Me, D, Kf, Me, D, D,
         (long long)Kf*Me, (long long)Kf*D, (long long)Me*D, Bb, 1,
         hei, Me, 1, nullptr, 0);

    // ---- down-Laplacian branch: d_0 *0^{-1} d_0^T *1 ----
    {
        long long tot = (long long)Bb * Me * D;
        scale_rows<<<(unsigned)((tot + 255) / 256), 256, 0, stream>>>(eV, he, y0, Me, tot);
    }
    gemm(d_0, y0, y1, Nv, D, Me, Nv, D, D,
         (long long)Me*Nv, (long long)Me*D, (long long)Nv*D, Bb, 1,
         hv, Nv, 2, nullptr, 0);
    gemm(d_0, y1, y0, Me, D, Nv, Nv, D, D,
         (long long)Me*Nv, (long long)Nv*D, (long long)Me*D, Bb, 0,
         nullptr, 0, 0, t2, (long long)Me*D);

    // ---- output projection ----
    gemm(y0, W_o, out, Bb*Me, D, D, D, D, D, 0,0,0, 1, 0, nullptr,0,0, nullptr,0);
}
